// Transformer_73581379715338
// MI455X (gfx1250) — compile-verified, matrix operand order checked by reading, not run on hardware
//
#include <hip/hip_runtime.h>
#include <hip/hip_bf16.h>
#include <math.h>
#include <stdint.h>

// Problem constants (match reference)
#define T_TOK   4096
#define D_MOD   512
#define H_HEADS 8
#define HW_DIM  64
#define FF_DIM  2048
#define N_LAYERS 6
#define EPS_LN  1e-12f

typedef __bf16 bf16_t;
typedef bf16_t v16bf __attribute__((ext_vector_type(16)));
typedef float  v8f   __attribute__((ext_vector_type(8)));

union FragU {
  v16bf v;
  unsigned int w[8];
  unsigned short h[16];
};

__device__ __forceinline__ v8f v8f_zero() {
  v8f z;
#pragma unroll
  for (int i = 0; i < 8; i++) z[i] = 0.0f;
  return z;
}

__device__ __forceinline__ v8f wmma_bf16(const FragU& a, const FragU& b, v8f c) {
  // D = A(16x32 bf16) * B(32x16 bf16) + C(16x16 f32)
  return __builtin_amdgcn_wmma_f32_16x16x32_bf16(false, a.v, false, b.v,
                                                 (short)0, c, false, false);
}

// ---------------------------------------------------------------------------
// CDNA5 async global->LDS copy (ASYNCcnt-tracked, bypasses VGPRs).
// LDS dest offset = low 32 bits of the generic shared-memory address
// (ISA 10.2: generic LDS addr[31:0] is the LDS offset).
// ---------------------------------------------------------------------------
__device__ __forceinline__ void async_b128(const bf16_t* g, bf16_t* l) {
  unsigned lds_off = (unsigned)(uintptr_t)l;
  unsigned long long ga = (unsigned long long)(uintptr_t)g;
  asm volatile("global_load_async_to_lds_b128 %0, %1, off"
               :: "v"(lds_off), "v"(ga) : "memory");
}
__device__ __forceinline__ void wait_async_le4() {
  asm volatile("s_wait_asynccnt 0x4" ::: "memory");
}
__device__ __forceinline__ void wait_async_le0() {
  asm volatile("s_wait_asynccnt 0x0" ::: "memory");
}

// ---------------------------------------------------------------------------
// f32 -> bf16 pack / init / copy-out helpers
// ---------------------------------------------------------------------------
__global__ void pack_bf16_kernel(const float* __restrict__ in,
                                 bf16_t* __restrict__ out, int n) {
  int i = blockIdx.x * blockDim.x + threadIdx.x;
  if (i < n) out[i] = (bf16_t)in[i];
}

__global__ void init_x_kernel(const float* __restrict__ h,
                              float* __restrict__ x,
                              bf16_t* __restrict__ xb, int n) {
  int i = blockIdx.x * blockDim.x + threadIdx.x;
  if (i < n) { float v = h[i]; x[i] = v; xb[i] = (bf16_t)v; }
}

__global__ void copy_out_kernel(const float* __restrict__ x,
                                float* __restrict__ o, int n) {
  int i = blockIdx.x * blockDim.x + threadIdx.x;
  if (i < n) o[i] = x[i];
}

// Per 16-row query tile: [first key (aligned down to 32), one-past-last key)
__global__ void seg_ranges_kernel(const int* __restrict__ seg,
                                  int* __restrict__ ranges, int ntiles) {
  int tile = blockIdx.x * blockDim.x + threadIdx.x;
  if (tile >= ntiles) return;
  int r0 = tile * 16;
  int s_lo = seg[r0], s_hi = seg[r0 + 15];
  int lo = 0, hi = T_TOK;
  while (lo < hi) { int mid = (lo + hi) >> 1; if (seg[mid] < s_lo) lo = mid + 1; else hi = mid; }
  int start = lo;
  lo = 0; hi = T_TOK;
  while (lo < hi) { int mid = (lo + hi) >> 1; if (seg[mid] <= s_hi) lo = mid + 1; else hi = mid; }
  ranges[2 * tile]     = start & ~31;
  ranges[2 * tile + 1] = lo;
}

// ---------------------------------------------------------------------------
// bf16 x bf16 -> f32 GEMM, 128x128 block tile, 256 threads (8 waves, wave tile
// 32x64 = 8 WMMA accums). Double-buffered LDS stages filled with CDNA5
// global_load_async_to_lds_b128, pipelined on ASYNCcnt.
// EPI: 0 = f32 out, 1 = bf16 out, 2 = bf16 swish(out)
// ---------------------------------------------------------------------------
template <int EPI>
__global__ __launch_bounds__(256, 1)
void gemm_bf16_kernel(const bf16_t* __restrict__ A, const bf16_t* __restrict__ B,
                      const float* __restrict__ bias,
                      float* __restrict__ Cf, bf16_t* __restrict__ Cb,
                      int M, int N, int K) {
  // A: row-major [m][k] (pad 40); B: row-major [k][n] (pad 136). Both padded to
  // keep every 16B async chunk 16B-aligned in LDS.
  __shared__ __align__(16) bf16_t As[2][128][40];
  __shared__ __align__(16) bf16_t Bs[2][32][136];

  const int tid   = threadIdx.x;
  const int lane  = tid & 31;
  const int wid   = tid >> 5;
  const int waveM = wid >> 1;        // 0..3
  const int waveN = wid & 1;         // 0..1
  const int ln16  = lane & 15;
  const int half8 = (lane >> 4) * 8; // 0 or 8

  const int mBase = blockIdx.y * 128;
  const int nBase = blockIdx.x * 128;

  v8f acc[2][4];
#pragma unroll
  for (int i = 0; i < 2; i++)
#pragma unroll
    for (int j = 0; j < 4; j++) acc[i][j] = v8f_zero();

  // Async stage issue: 4 b128 per thread (2 for A 128x32, 2 for B 32x128)
  auto issue_stage = [&](int s) {
    const int st = s & 1;
    const int k0 = s * 32;
#pragma unroll
    for (int t = 0; t < 2; t++) {
      const int c  = tid + t * 256;      // 0..511
      const int r  = c >> 2;             // 0..127
      const int c8 = (c & 3) * 8;        // 0,8,16,24
      async_b128(A + (size_t)(mBase + r) * K + k0 + c8, &As[st][r][c8]);
    }
#pragma unroll
    for (int t = 0; t < 2; t++) {
      const int c  = tid + t * 256;      // 0..511
      const int k  = c >> 4;             // 0..31
      const int n8 = (c & 15) * 8;       // 0..120
      async_b128(B + (size_t)(k0 + k) * N + nBase + n8, &Bs[st][k][n8]);
    }
  };

  const int S = K >> 5;
  issue_stage(0);

  for (int s = 0; s < S; s++) {
    if (s + 1 < S) {
      issue_stage(s + 1);   // prefetch next stage into the other buffer
      wait_async_le4();     // oldest 4 (= this stage) have landed
    } else {
      wait_async_le0();
    }
    __syncthreads();        // all waves' stage data visible

    const int st = s & 1;
    FragU afr[2], bfr[4];
#pragma unroll
    for (int mt = 0; mt < 2; mt++) {
      const int m = waveM * 32 + mt * 16 + ln16;
#pragma unroll
      for (int j = 0; j < 8; j++) {
        const int kw = half8 + 2 * j + ((j >= 4) ? 8 : 0);
        afr[mt].w[j] = *(const unsigned int*)&As[st][m][kw];
      }
    }
#pragma unroll
    for (int nt = 0; nt < 4; nt++) {
      const int n = waveN * 64 + nt * 16 + ln16;
#pragma unroll
      for (int j = 0; j < 8; j++) {
        const int kw = half8 + 2 * j + ((j >= 4) ? 8 : 0);
        bfr[nt].h[2 * j]     = *(const unsigned short*)&Bs[st][kw][n];
        bfr[nt].h[2 * j + 1] = *(const unsigned short*)&Bs[st][kw + 1][n];
      }
    }
#pragma unroll
    for (int mt = 0; mt < 2; mt++)
#pragma unroll
      for (int nt = 0; nt < 4; nt++)
        acc[mt][nt] = wmma_bf16(afr[mt], bfr[nt], acc[mt][nt]);
    __syncthreads();        // WAR: buffer may be refilled next iteration
  }

  // Epilogue
#pragma unroll
  for (int mt = 0; mt < 2; mt++) {
#pragma unroll
    for (int nt = 0; nt < 4; nt++) {
      const int col = nBase + waveN * 64 + nt * 16 + ln16;
      const float bia = bias ? bias[col] : 0.0f;
#pragma unroll
      for (int i = 0; i < 8; i++) {
        const int row = mBase + waveM * 32 + mt * 16 + half8 + i;
        float v = acc[mt][nt][i] + bia;
        if (EPI == 0) {
          Cf[(size_t)row * N + col] = v;
        } else if (EPI == 1) {
          Cb[(size_t)row * N + col] = (bf16_t)v;
        } else {
          float sw = v / (1.0f + __expf(-v));
          Cb[(size_t)row * N + col] = (bf16_t)sw;
        }
      }
    }
  }
}

// ---------------------------------------------------------------------------
// Segment-masked flash attention. One wave per (16-query tile, head).
// V tile staged with async-to-LDS, overlapped with QK^T + softmax.
// Emits: ab = bf16( softmax(QK^T/8, seg-mask) V  +  c )
// ---------------------------------------------------------------------------
__global__ __launch_bounds__(32, 1)
void attn_kernel(const bf16_t* __restrict__ qb, const bf16_t* __restrict__ kb,
                 const bf16_t* __restrict__ vb, const float* __restrict__ cf,
                 const int* __restrict__ seg, const int* __restrict__ ranges,
                 bf16_t* __restrict__ ab) {
  __shared__ __align__(16) bf16_t Ps[16][34];   // P tile (queries x 32 keys)
  __shared__ __align__(16) bf16_t Vs[32][72];   // V tile (keys x 64 dims)

  const int lane  = threadIdx.x & 31;
  const int ln16  = lane & 15;
  const int half8 = (lane >> 4) * 8;

  const int qt    = blockIdx.x;
  const int hh    = blockIdx.y;
  const int qrow0 = qt * 16;
  const int dbase = hh * HW_DIM;

  const int jlo = ranges[2 * qt];
  const int jhi = ranges[2 * qt + 1];

  // Q: 16x64 -> two K=32 A-fragments, gathered straight from global
  FragU aq[2];
  {
    const bf16_t* qptr = qb + (size_t)(qrow0 + ln16) * D_MOD + dbase;
#pragma unroll
    for (int f = 0; f < 2; f++)
#pragma unroll
      for (int j = 0; j < 8; j++) {
        const int kw = f * 32 + half8 + 2 * j + ((j >= 4) ? 8 : 0);
        aq[f].w[j] = *(const unsigned int*)(qptr + kw);
      }
  }

  int qseg[8];
#pragma unroll
  for (int i = 0; i < 8; i++) qseg[i] = seg[qrow0 + half8 + i];

  v8f acc_o[4];
#pragma unroll
  for (int nc = 0; nc < 4; nc++) acc_o[nc] = v8f_zero();
  float rmax[8], rsum[8];
#pragma unroll
  for (int i = 0; i < 8; i++) { rmax[i] = -3.0e38f; rsum[i] = 0.0f; }

  const float scale = 0.125f;  // 1/sqrt(64)

  for (int j0 = jlo; j0 < jhi; j0 += 32) {
    // Kick off async V-tile copy (32 keys x 64 dims); lands while we do QK^T.
    // Previous iteration's __syncthreads (DS-count wait) covers the WAR hazard.
    {
      const bf16_t* vptr = vb + (size_t)(j0 + lane) * D_MOD + dbase;
#pragma unroll
      for (int c = 0; c < 8; c++)
        async_b128(vptr + c * 8, &Vs[lane][c * 8]);
    }

    float p0[8], p1[8];
    {
      v8f s[2];
#pragma unroll
      for (int g = 0; g < 2; g++) {
        const int key = j0 + g * 16 + ln16;
        const bf16_t* kptr = kb + (size_t)key * D_MOD + dbase;
        FragU bk0, bk1;
#pragma unroll
        for (int j = 0; j < 8; j++) {
          const int kw = half8 + 2 * j + ((j >= 4) ? 8 : 0);
          bk0.w[j] = *(const unsigned int*)(kptr + kw);
          bk1.w[j] = *(const unsigned int*)(kptr + 32 + kw);
        }
        v8f z = v8f_zero();
        z = wmma_bf16(aq[0], bk0, z);
        z = wmma_bf16(aq[1], bk1, z);
        s[g] = z;
      }
      const int ks0 = seg[j0 + ln16];
      const int ks1 = seg[j0 + 16 + ln16];
#pragma unroll
      for (int i = 0; i < 8; i++) {
        p0[i] = (ks0 == qseg[i]) ? s[0][i] * scale : -3.0e38f;
        p1[i] = (ks1 == qseg[i]) ? s[1][i] * scale : -3.0e38f;
      }
    }

    // Online softmax: row = query, spread over 16 lanes of each half-wave
#pragma unroll
    for (int i = 0; i < 8; i++) {
      float t = fmaxf(p0[i], p1[i]);
#pragma unroll
      for (int m = 1; m < 16; m <<= 1) t = fmaxf(t, __shfl_xor(t, m, 32));
      float mnew  = fmaxf(rmax[i], t);
      float alpha = (rmax[i] > -1.0e37f) ? __expf(rmax[i] - mnew) : 0.0f;
      float e0 = (p0[i] > -1.0e37f) ? __expf(p0[i] - mnew) : 0.0f;
      float e1 = (p1[i] > -1.0e37f) ? __expf(p1[i] - mnew) : 0.0f;
      float rs = e0 + e1;
#pragma unroll
      for (int m = 1; m < 16; m <<= 1) rs += __shfl_xor(rs, m, 32);
      rsum[i] = rsum[i] * alpha + rs;
      rmax[i] = mnew;
#pragma unroll
      for (int nc = 0; nc < 4; nc++) acc_o[nc][i] *= alpha;
      p0[i] = e0; p1[i] = e1;
    }

    // P (D-layout) -> LDS -> A-layout
#pragma unroll
    for (int i = 0; i < 8; i++) {
      Ps[half8 + i][ln16]      = (bf16_t)p0[i];
      Ps[half8 + i][16 + ln16] = (bf16_t)p1[i];
    }
    wait_async_le0();   // V tile landed
    __syncthreads();

    FragU ap;
#pragma unroll
    for (int j = 0; j < 8; j++) {
      const int kw = half8 + 2 * j + ((j >= 4) ? 8 : 0);
      ap.w[j] = *(const unsigned int*)&Ps[ln16][kw];
    }
#pragma unroll
    for (int nc = 0; nc < 4; nc++) {
      FragU bvf;
      const int n = nc * 16 + ln16;
#pragma unroll
      for (int j = 0; j < 8; j++) {
        const int kw = half8 + 2 * j + ((j >= 4) ? 8 : 0);
        bvf.h[2 * j]     = *(const unsigned short*)&Vs[kw][n];
        bvf.h[2 * j + 1] = *(const unsigned short*)&Vs[kw + 1][n];
      }
      acc_o[nc] = wmma_bf16(ap, bvf, acc_o[nc]);
    }
    __syncthreads();
  }

  // out = acc/rowsum + c
#pragma unroll
  for (int nc = 0; nc < 4; nc++) {
    const int col = dbase + nc * 16 + ln16;
#pragma unroll
    for (int i = 0; i < 8; i++) {
      const int row = qrow0 + half8 + i;
      float v = acc_o[nc][i] / rsum[i] + cf[(size_t)row * D_MOD + col];
      ab[(size_t)row * D_MOD + col] = (bf16_t)v;
    }
  }
}

// ---------------------------------------------------------------------------
// x = LN(res + delta) (TF-style: biased var, eps inside sqrt); also emits bf16
// ---------------------------------------------------------------------------
__global__ __launch_bounds__(256, 1)
void ln_kernel(const float* __restrict__ res, const float* __restrict__ delta,
               const float* __restrict__ gamma, const float* __restrict__ beta,
               float* __restrict__ xout, bf16_t* __restrict__ xbout) {
  __shared__ float sh[16];
  const int row = blockIdx.x;
  const int tid = threadIdx.x;
  const float* r = res + (size_t)row * D_MOD;
  const float* d = delta + (size_t)row * D_MOD;
  float v0 = r[tid] + d[tid];
  float v1 = r[tid + 256] + d[tid + 256];
  float s1 = v0 + v1;
  float s2 = v0 * v0 + v1 * v1;
#pragma unroll
  for (int m = 1; m < 32; m <<= 1) {
    s1 += __shfl_xor(s1, m, 32);
    s2 += __shfl_xor(s2, m, 32);
  }
  if ((tid & 31) == 0) { sh[tid >> 5] = s1; sh[8 + (tid >> 5)] = s2; }
  __syncthreads();
  float ts1 = 0.0f, ts2 = 0.0f;
#pragma unroll
  for (int i = 0; i < 8; i++) { ts1 += sh[i]; ts2 += sh[8 + i]; }
  const float mean = ts1 * (1.0f / D_MOD);
  const float var  = ts2 * (1.0f / D_MOD) - mean * mean;
  const float inv  = rsqrtf(var + EPS_LN);
  float y0 = gamma[tid] * (v0 - mean) * inv + beta[tid];
  float y1 = gamma[tid + 256] * (v1 - mean) * inv + beta[tid + 256];
  float* xo = xout + (size_t)row * D_MOD;
  bf16_t* xbo = xbout + (size_t)row * D_MOD;
  xo[tid] = y0;          xo[tid + 256] = y1;
  xbo[tid] = (bf16_t)y0; xbo[tid + 256] = (bf16_t)y1;
}

// ---------------------------------------------------------------------------
// Host orchestration
// ---------------------------------------------------------------------------
extern "C" void kernel_launch(void* const* d_in, const int* in_sizes, int n_in,
                              void* d_out, int out_size, void* d_ws, size_t ws_size,
                              hipStream_t stream) {
  const float* h   = (const float*)d_in[0];
  const int*   seg = (const int*)d_in[1];
  const float* Wq  = (const float*)d_in[2];  const float* bq  = (const float*)d_in[3];
  const float* Wk  = (const float*)d_in[4];  const float* bk  = (const float*)d_in[5];
  const float* Wv  = (const float*)d_in[6];  const float* bvp = (const float*)d_in[7];
  const float* Wc  = (const float*)d_in[8];  const float* bc  = (const float*)d_in[9];
  const float* Wp  = (const float*)d_in[10]; const float* bp  = (const float*)d_in[11];
  const float* g1  = (const float*)d_in[12]; const float* b1  = (const float*)d_in[13];
  const float* g2  = (const float*)d_in[14]; const float* b2  = (const float*)d_in[15];
  const float* W1  = (const float*)d_in[16]; const float* fb1 = (const float*)d_in[17];
  const float* W2  = (const float*)d_in[18]; const float* fb2 = (const float*)d_in[19];

  char* ws = (char*)d_ws;
  size_t off = 0;
  auto alloc = [&](size_t bytes) -> void* {
    void* p = ws + off;
    off = (off + bytes + 255) & ~(size_t)255;
    return p;
  };

  const size_t TD = (size_t)T_TOK * D_MOD;
  const size_t TF = (size_t)T_TOK * FF_DIM;
  const size_t DD = (size_t)D_MOD * D_MOD;
  const size_t DF = (size_t)D_MOD * FF_DIM;

  float*  x    = (float*) alloc(TD * 4);
  bf16_t* xb   = (bf16_t*)alloc(TD * 2);
  bf16_t* qbuf = (bf16_t*)alloc(TD * 2);
  bf16_t* kbuf = (bf16_t*)alloc(TD * 2);
  bf16_t* vbuf = (bf16_t*)alloc(TD * 2);
  float*  cbuf = (float*) alloc(TD * 4);
  bf16_t* abuf = (bf16_t*)alloc(TD * 2);
  float*  proj = (float*) alloc(TD * 4);
  bf16_t* ffb  = (bf16_t*)alloc(TF * 2);
  float*  f2   = (float*) alloc(TD * 4);
  bf16_t* Wqb  = (bf16_t*)alloc(DD * 2);
  bf16_t* Wkb  = (bf16_t*)alloc(DD * 2);
  bf16_t* Wvb  = (bf16_t*)alloc(DD * 2);
  bf16_t* Wcb  = (bf16_t*)alloc(DD * 2);
  bf16_t* Wpb  = (bf16_t*)alloc(DD * 2);
  bf16_t* W1b  = (bf16_t*)alloc(DF * 2);
  bf16_t* W2b  = (bf16_t*)alloc(DF * 2);
  int*    rng  = (int*)   alloc((T_TOK / 16) * 2 * sizeof(int));

  const int nDD = (int)DD, nDF = (int)DF, nTD = (int)TD;
  pack_bf16_kernel<<<(nDD + 255) / 256, 256, 0, stream>>>(Wq, Wqb, nDD);
  pack_bf16_kernel<<<(nDD + 255) / 256, 256, 0, stream>>>(Wk, Wkb, nDD);
  pack_bf16_kernel<<<(nDD + 255) / 256, 256, 0, stream>>>(Wv, Wvb, nDD);
  pack_bf16_kernel<<<(nDD + 255) / 256, 256, 0, stream>>>(Wc, Wcb, nDD);
  pack_bf16_kernel<<<(nDD + 255) / 256, 256, 0, stream>>>(Wp, Wpb, nDD);
  pack_bf16_kernel<<<(nDF + 255) / 256, 256, 0, stream>>>(W1, W1b, nDF);
  pack_bf16_kernel<<<(nDF + 255) / 256, 256, 0, stream>>>(W2, W2b, nDF);
  seg_ranges_kernel<<<1, 256, 0, stream>>>(seg, rng, T_TOK / 16);
  init_x_kernel<<<(nTD + 255) / 256, 256, 0, stream>>>(h, x, xb, nTD);

  dim3 gD(D_MOD / 128, T_TOK / 128);   // GEMMs with N = 512
  dim3 gF(FF_DIM / 128, T_TOK / 128);  // GEMM with N = 2048
  dim3 gA(T_TOK / 16, H_HEADS);

  for (int l = 0; l < N_LAYERS; ++l) {
    gemm_bf16_kernel<1><<<gD, 256, 0, stream>>>(xb, Wqb, bq,  nullptr, qbuf, T_TOK, D_MOD, D_MOD);
    gemm_bf16_kernel<1><<<gD, 256, 0, stream>>>(xb, Wkb, bk,  nullptr, kbuf, T_TOK, D_MOD, D_MOD);
    gemm_bf16_kernel<1><<<gD, 256, 0, stream>>>(xb, Wvb, bvp, nullptr, vbuf, T_TOK, D_MOD, D_MOD);
    gemm_bf16_kernel<0><<<gD, 256, 0, stream>>>(xb, Wcb, bc,  cbuf, nullptr, T_TOK, D_MOD, D_MOD);
    attn_kernel<<<gA, 32, 0, stream>>>(qbuf, kbuf, vbuf, cbuf, seg, rng, abuf);
    gemm_bf16_kernel<0><<<gD, 256, 0, stream>>>(abuf, Wpb, bp, proj, nullptr, T_TOK, D_MOD, D_MOD);
    ln_kernel<<<T_TOK, 256, 0, stream>>>(x, proj, g1, b1, x, xb);
    gemm_bf16_kernel<2><<<gF, 256, 0, stream>>>(xb, W1b, fb1, nullptr, ffb, T_TOK, FF_DIM, D_MOD);
    gemm_bf16_kernel<0><<<gD, 256, 0, stream>>>(ffb, W2b, fb2, f2, nullptr, T_TOK, D_MOD, FF_DIM);
    ln_kernel<<<T_TOK, 256, 0, stream>>>(x, f2, g2, b2, x, xb);
  }
  copy_out_kernel<<<(nTD + 255) / 256, 256, 0, stream>>>(x, (float*)d_out, nTD);
}
